// GINBlock_90898687852683
// MI455X (gfx1250) — matrix operand order repaired and checked by reading, not compiled, not run on hardware
//
#include <hip/hip_runtime.h>
#include <hip/hip_bf16.h>

typedef _Float16 v8h  __attribute__((ext_vector_type(8)));
typedef _Float16 v16h __attribute__((ext_vector_type(16)));
typedef float    v8f  __attribute__((ext_vector_type(8)));

#define D_FEAT 128
#define BN_EPS 1e-5f

// Guaranteed CDNA5 hardware f32 atomic, device scope (RMW resolves in L2).
__device__ __forceinline__ void atomic_add_f32(float* p, float v) {
    unsigned long long addr = (unsigned long long)p;
    asm volatile("global_atomic_add_f32 %0, %1, off scope:SCOPE_DEV"
                 :: "v"(addr), "v"(v)
                 : "memory");
}

// ---------------------------------------------------------------- zero
__global__ void zero_kernel(float* __restrict__ p, long long n) {
    long long t = (long long)blockIdx.x * blockDim.x + threadIdx.x;
    if (t < n) p[t] = 0.0f;
}

// ---------------------------------------------------------------- weight split: W -> hi/lo f16
__global__ void wsplit_kernel(const float* __restrict__ W,
                              _Float16* __restrict__ Wh, _Float16* __restrict__ Wl, int n) {
    int i = blockIdx.x * blockDim.x + threadIdx.x;
    if (i >= n) return;
    float w = W[i];
    _Float16 h = (_Float16)w;
    Wh[i] = h;
    Wl[i] = (_Float16)(w - (float)h);
}

// ---------------------------------------------------------------- scatter-add aggregation
// one wave (32 lanes) per edge; each lane moves a float4 chunk (32*4 = 128 floats)
__global__ void scatter_kernel(const float* __restrict__ x,
                               const long long* __restrict__ ei,
                               float* __restrict__ agg, long long E) {
    long long t = (long long)blockIdx.x * blockDim.x + threadIdx.x;
    long long e = t >> 5;
    if (e >= E) return;
    int lane = (int)(t & 31);
    long long src = ei[e];
    long long dst = ei[E + e];
    float4 v = ((const float4*)(x + src * D_FEAT))[lane];
    float* d = agg + dst * D_FEAT + lane * 4;
    atomic_add_f32(d + 0, v.x);
    atomic_add_f32(d + 1, v.y);
    atomic_add_f32(d + 2, v.z);
    atomic_add_f32(d + 3, v.w);
}

// ---------------------------------------------------------------- WMMA GEMM (split-f16, ~f32 accurate)
// out[n][j] = sum_k A[n][k] * W[j][k] + bias[j]   (A @ W^T + b)
// A and W are split as x = hi + lo (f16 each); product accumulated in f32 via
// 3 WMMAs per K-step: Ah*Bh + Al*Bh + Ah*Bl  (Al*Bl term ~2^-22, dropped).
// MODE 0: A = (1+eps)*X0 + X1              (GIN combine; p0 = &eps)
// MODE 1: A = relu(scale[c]*X0 + shift[c]) (BN+ReLU affine on the fly)
// Epilogue also accumulates per-column sum / sumsq for the following BatchNorm.
template <int MODE>
__global__ __launch_bounds__(256) void gin_gemm_kernel(
    const float* __restrict__ X0, const float* __restrict__ X1,
    const float* __restrict__ p0, const float* __restrict__ p1,
    const _Float16* __restrict__ Wh, const _Float16* __restrict__ Wl,
    const float* __restrict__ bias,
    float* __restrict__ Hout, float* __restrict__ colsum, float* __restrict__ colsq) {

    __shared__ __align__(16) _Float16 Ah[16][D_FEAT];  // 4 KB hi plane
    __shared__ __align__(16) _Float16 Al[16][D_FEAT];  // 4 KB lo plane

    const int tile = blockIdx.x;   // 16-row stripe of nodes
    const int tid  = threadIdx.x;  // 256 threads = 8 waves

    // ---- stage A tile into LDS as split f16 (each thread: 8 consecutive elems in one row)
    {
        int idx  = tid * 8;
        int r    = idx >> 7;
        int c0   = idx & (D_FEAT - 1);
        long long grow = (long long)tile * 16 + r;
        const float4* q0 = (const float4*)(X0 + grow * D_FEAT + c0);
        float4 u0 = q0[0], u1 = q0[1];
        float vals[8];
        if (MODE == 0) {
            float s = 1.0f + p0[0];
            const float4* q1 = (const float4*)(X1 + grow * D_FEAT + c0);
            float4 v0 = q1[0], v1 = q1[1];
            vals[0] = s * u0.x + v0.x; vals[1] = s * u0.y + v0.y;
            vals[2] = s * u0.z + v0.z; vals[3] = s * u0.w + v0.w;
            vals[4] = s * u1.x + v1.x; vals[5] = s * u1.y + v1.y;
            vals[6] = s * u1.z + v1.z; vals[7] = s * u1.w + v1.w;
        } else {
            const float* sc = p0 + c0;
            const float* sh = p1 + c0;
            vals[0] = fmaxf(fmaf(sc[0], u0.x, sh[0]), 0.0f);
            vals[1] = fmaxf(fmaf(sc[1], u0.y, sh[1]), 0.0f);
            vals[2] = fmaxf(fmaf(sc[2], u0.z, sh[2]), 0.0f);
            vals[3] = fmaxf(fmaf(sc[3], u0.w, sh[3]), 0.0f);
            vals[4] = fmaxf(fmaf(sc[4], u1.x, sh[4]), 0.0f);
            vals[5] = fmaxf(fmaf(sc[5], u1.y, sh[5]), 0.0f);
            vals[6] = fmaxf(fmaf(sc[6], u1.z, sh[6]), 0.0f);
            vals[7] = fmaxf(fmaf(sc[7], u1.w, sh[7]), 0.0f);
        }
        _Float16* dh = &Ah[r][c0];
        _Float16* dl = &Al[r][c0];
#pragma unroll
        for (int k = 0; k < 8; ++k) {
            _Float16 h = (_Float16)vals[k];
            dh[k] = h;
            dl[k] = (_Float16)(vals[k] - (float)h);
        }
    }
    __syncthreads();

    const int wave = tid >> 5;
    const int lane = tid & 31;
    const int j0   = wave * 16;            // this wave's 16 output columns
    const int j    = j0 + (lane & 15);     // this lane's column (B/C layout: N = lane)
    const int hi   = lane & 16;            // 0 for lanes 0-15, 16 for lanes 16-31
    const int m    = lane & 15;            // A layout: M = lane
    const long long jrow = (long long)j * D_FEAT;

    v8f acc = {0.f, 0.f, 0.f, 0.f, 0.f, 0.f, 0.f, 0.f};

#pragma unroll
    for (int ks = 0; ks < 4; ++ks) {       // K = 128 in 4 steps of 32
        const int k0 = ks * 32;
        // A fragments (16x32): lane = M row, elem i -> K = k0 + hi + i (lanes 0-15: K 0-15; lanes 16-31: K 16-31)
        v16h a_hi = *(const v16h*)&Ah[m][k0 + hi];
        v16h a_lo = *(const v16h*)&Al[m][k0 + hi];
        // B fragments (32x16): lane = N col, elem i -> K = k0 + hi + i ; B[k][j] = W[j*128+k]
        v16h b_hi = *(const v16h*)(Wh + jrow + k0 + hi);
        v16h b_lo = *(const v16h*)(Wl + jrow + k0 + hi);

        acc = __builtin_amdgcn_wmma_f32_16x16x32_f16(false, a_hi, false, b_hi,
                                                     (short)0, acc, false, false);
        acc = __builtin_amdgcn_wmma_f32_16x16x32_f16(false, a_lo, false, b_hi,
                                                     (short)0, acc, false, false);
        acc = __builtin_amdgcn_wmma_f32_16x16x32_f16(false, a_hi, false, b_lo,
                                                     (short)0, acc, false, false);
    }

    // ---- epilogue: add bias, store f32, accumulate BN statistics
    const float bj = bias[j];
    const int mbase = hi >> 1;             // C layout: lanes 0-15 -> rows r, lanes 16-31 -> r+8
    float lsum = 0.f, lsq = 0.f;
#pragma unroll
    for (int r = 0; r < 8; ++r) {
        long long row = (long long)tile * 16 + mbase + r;
        float v = acc[r] + bj;
        Hout[row * D_FEAT + j] = v;
        lsum += v;
        lsq  += v * v;
    }
    atomic_add_f32(&colsum[j], lsum);
    atomic_add_f32(&colsq[j], lsq);
}

// ---------------------------------------------------------------- BN finalize: per-column affine
__global__ void bn_finalize_kernel(const float* __restrict__ sum, const float* __restrict__ sq,
                                   const float* __restrict__ gamma, const float* __restrict__ beta,
                                   float invN, float* __restrict__ a, float* __restrict__ c) {
    int jj = threadIdx.x;  // 128 threads
    float mu  = sum[jj] * invN;
    float var = sq[jj] * invN - mu * mu;
    float s   = gamma[jj] * rsqrtf(var + BN_EPS);
    a[jj] = s;
    c[jj] = beta[jj] - mu * s;
}

// ---------------------------------------------------------------- final BN + ReLU
__global__ void bn_relu_out_kernel(const float* __restrict__ h, const float* __restrict__ a,
                                   const float* __restrict__ c, float* __restrict__ out,
                                   long long total4) {
    long long t = (long long)blockIdx.x * blockDim.x + threadIdx.x;
    if (t >= total4) return;
    int c4 = (int)(t & 31) * 4;  // 32 float4 per 128-col row
    float4 v = ((const float4*)h)[t];
    float4 o;
    o.x = fmaxf(fmaf(a[c4 + 0], v.x, c[c4 + 0]), 0.0f);
    o.y = fmaxf(fmaf(a[c4 + 1], v.y, c[c4 + 1]), 0.0f);
    o.z = fmaxf(fmaf(a[c4 + 2], v.z, c[c4 + 2]), 0.0f);
    o.w = fmaxf(fmaf(a[c4 + 3], v.w, c[c4 + 3]), 0.0f);
    ((float4*)out)[t] = o;
}

// ================================================================ launch
extern "C" void kernel_launch(void* const* d_in, const int* in_sizes, int n_in,
                              void* d_out, int out_size, void* d_ws, size_t ws_size,
                              hipStream_t stream) {
    const float*      x   = (const float*)d_in[0];
    const long long*  ei  = (const long long*)d_in[1];
    const float*      eps = (const float*)d_in[2];
    const float*      W1  = (const float*)d_in[3];
    const float*      b1  = (const float*)d_in[4];
    const float*      g1  = (const float*)d_in[5];
    const float*      be1 = (const float*)d_in[6];
    const float*      W2  = (const float*)d_in[7];
    const float*      b2  = (const float*)d_in[8];
    const float*      g2  = (const float*)d_in[9];
    const float*      be2 = (const float*)d_in[10];
    float*            out = (float*)d_out;

    const long long N  = (long long)in_sizes[0] / D_FEAT;  // 50000 (multiple of 16)
    const long long E  = (long long)in_sizes[1] / 2;       // 800000
    const long long ND = N * D_FEAT;
    const int       WN = D_FEAT * D_FEAT;                  // 16384 weights per layer

    float* ws   = (float*)d_ws;
    float* bufA = ws;             // agg, reused as h2 after GEMM1 consumes it
    float* bufB = ws + ND;        // h1
    float* st   = ws + 2 * ND;    // stats + affine params (1024 floats)
    float* sum1 = st,       *sq1 = st + 128;
    float* sum2 = st + 256, *sq2 = st + 384;
    float* a1   = st + 512, *c1  = st + 640;
    float* a2   = st + 768, *c2  = st + 896;
    // split-f16 weight tables (each 16384 f16 = 8192 float slots; 32B aligned)
    _Float16* W1h = (_Float16*)(st + 1024);
    _Float16* W1l = (_Float16*)(st + 1024 + 8192);
    _Float16* W2h = (_Float16*)(st + 1024 + 16384);
    _Float16* W2l = (_Float16*)(st + 1024 + 24576);

    const float invN = 1.0f / (float)N;

    // 1) zero aggregation buffer + stat accumulators; pre-split weights to f16 hi/lo
    {
        int blocks = (int)((ND + 255) / 256);
        zero_kernel<<<blocks, 256, 0, stream>>>(bufA, ND);
        zero_kernel<<<4, 256, 0, stream>>>(st, 1024);
        wsplit_kernel<<<WN / 256, 256, 0, stream>>>(W1, W1h, W1l, WN);
        wsplit_kernel<<<WN / 256, 256, 0, stream>>>(W2, W2h, W2l, WN);
    }
    // 2) scatter-add: agg[dst] += x[src]   (hw global_atomic_add_f32, L2-resident target)
    {
        long long threads = E * 32;
        int blocks = (int)((threads + 255) / 256);
        scatter_kernel<<<blocks, 256, 0, stream>>>(x, ei, bufA, E);
    }
    // 3) h1 = ((1+eps)*x + agg) @ W1^T + b1 ; accumulate BN1 stats
    gin_gemm_kernel<0><<<(int)(N / 16), 256, 0, stream>>>(
        x, bufA, eps, nullptr, W1h, W1l, b1, bufB, sum1, sq1);
    // 4) BN1 affine
    bn_finalize_kernel<<<1, 128, 0, stream>>>(sum1, sq1, g1, be1, invN, a1, c1);
    // 5) h2 = relu(a1*h1+c1) @ W2^T + b2 ; accumulate BN2 stats (writes into bufA)
    gin_gemm_kernel<1><<<(int)(N / 16), 256, 0, stream>>>(
        bufB, nullptr, a1, c1, W2h, W2l, b2, bufA, sum2, sq2);
    // 6) BN2 affine
    bn_finalize_kernel<<<1, 128, 0, stream>>>(sum2, sq2, g2, be2, invN, a2, c2);
    // 7) out = relu(a2*h2 + c2)
    {
        long long total4 = ND / 4;
        int blocks = (int)((total4 + 255) / 256);
        bn_relu_out_kernel<<<blocks, 256, 0, stream>>>(bufA, a2, c2, out, total4);
    }
}
